// DCN_64733747085505
// MI455X (gfx1250) — matrix operand order repaired and checked by reading, not compiled, not run on hardware
//
#include <hip/hip_runtime.h>
#include <math.h>

typedef __attribute__((ext_vector_type(16))) _Float16 v16h;
typedef __attribute__((ext_vector_type(8)))  _Float16 v8h;
typedef __attribute__((ext_vector_type(8)))  float    v8f;

#define OHT 256
#define OWD 256
#define NPIX (2 * OHT * OWD)   // 131072 output pixels
#define IHT 512
#define IWD 512
#define KCONV 576              // Cin(64) * 3 * 3

// ---------------- Kernel 0: weight prep (f32 -> f16, transposes) ----------------
__global__ __launch_bounds__(256)
void k_prep(const float* __restrict__ conv_w, const float* __restrict__ w_in,
            const float* __restrict__ w_out, _Float16* __restrict__ cw16,
            _Float16* __restrict__ win_t, _Float16* __restrict__ wout_t) {
  int idx = blockIdx.x * 256 + threadIdx.x;
  if (idx < 64 * KCONV) {
    // (O,I,3,3) flattens to [cout][ci*9+kh*3+kw] == im2col K order
    cw16[idx] = (_Float16)conv_w[idx];
  } else if (idx < 64 * KCONV + 4096) {
    int t = idx - 64 * KCONV; int j = t >> 6, c = t & 63;
    win_t[t] = (_Float16)w_in[c * 64 + j];       // win_t[j][c] = w_in[c][j]
  } else if (idx < 64 * KCONV + 8192) {
    int t = idx - 64 * KCONV - 4096; int j = t >> 6, c = t & 63;
    wout_t[t] = (_Float16)w_out[c * 64 + j];
  }
}

// ---------------- Kernel 1: stride-2 conv + ReLU via im2col WMMA GEMM ----------------
// Block = 128 thr (4 waves). Tile: 16 pixels (run along OW) x 64 out channels.
__global__ __launch_bounds__(128)
void k_conv(const float* __restrict__ x, const _Float16* __restrict__ cw16,
            float* __restrict__ y32, _Float16* __restrict__ y16) {
  __shared__ __align__(16) _Float16 Alds[16][584];   // 16 x 576 im2col rows (+pad)
  int b   = blockIdx.x;
  int ow0 = (b & 15) << 4;
  int oh  = (b >> 4) & 255;
  int n   = b >> 12;
  // im2col fill (uniform trip count: 9216/128 = 72, no divergence)
  for (int idx = threadIdx.x; idx < 16 * KCONV; idx += 128) {
    int pix = idx / KCONV;
    int k   = idx - pix * KCONV;
    int ci  = k / 9;
    int r   = k - ci * 9;
    int kh  = r / 3;
    int kw  = r - kh * 3;
    int ih = (oh << 1) - 1 + kh;
    int iw = ((ow0 + pix) << 1) - 1 + kw;
    float v = 0.f;
    if ((unsigned)ih < IHT && (unsigned)iw < IWD)
      v = x[(((size_t)n * 64 + ci) * IHT + ih) * IWD + iw];
    Alds[pix][k] = (_Float16)v;
  }
  __syncthreads();
  int lane = threadIdx.x & 31;
  int wv   = threadIdx.x >> 5;             // N-tile (16 output channels)
  int col  = (wv << 4) + (lane & 15);
  int arow = lane & 15;
  int aofs = (lane & 16) ? 16 : 0;         // A layout: 16 contiguous K per half-wave
  int bofs = (lane & 16) ? 16 : 0;         // B layout: 16 contiguous K per half-wave
  v8f acc = {};
  #pragma unroll
  for (int kt = 0; kt < 18; ++kt) {        // K = 576 = 18 x 32
    int kb = kt << 5;
    const _Float16* ap = &Alds[arow][kb + aofs];
    v8h lo = *(const v8h*)ap;
    v8h hi = *(const v8h*)(ap + 8);
    v16h a;
    #pragma unroll
    for (int i = 0; i < 8; ++i) { a[i] = lo[i]; a[i + 8] = hi[i]; }
    v16h bb = *(const v16h*)(cw16 + (size_t)col * KCONV + kb + bofs);
    acc = __builtin_amdgcn_wmma_f32_16x16x32_f16(false, a, false, bb,
                                                 (short)0, acc, false, false);
  }
  #pragma unroll
  for (int i = 0; i < 8; ++i) {            // C/D: VGPR i = row i (lanes<16) / i+8
    int prow = (lane & 16) ? (i + 8) : i;
    size_t idx = ((((size_t)n * OHT + oh) * OWD) + ow0 + prow) * 64 + col;
    float v = fmaxf(acc[i], 0.f);          // ReLU
    y32[idx] = v;
    y16[idx] = (_Float16)v;
  }
}

// ---------------- shared 16-row x K=64 x N=64 WMMA GEMM fragment ----------------
__device__ inline v8f gemm16x64(const _Float16* __restrict__ A,
                                const _Float16* __restrict__ Bt,
                                int pix0, int lane, int wv) {
  int arow = pix0 + (lane & 15);
  int col  = (wv << 4) + (lane & 15);
  int aofs = (lane & 16) ? 16 : 0;
  int bofs = (lane & 16) ? 16 : 0;
  v8f acc = {};
  #pragma unroll
  for (int kt = 0; kt < 2; ++kt) {
    int kb = kt << 5;
    const _Float16* ap = A + (size_t)arow * 64 + kb + aofs;
    v8h lo = *(const v8h*)ap;
    v8h hi = *(const v8h*)(ap + 8);
    v16h a;
    #pragma unroll
    for (int i = 0; i < 8; ++i) { a[i] = lo[i]; a[i + 8] = hi[i]; }
    v16h bb = *(const v16h*)(Bt + (size_t)col * 64 + kb + bofs);
    acc = __builtin_amdgcn_wmma_f32_16x16x32_f16(false, a, false, bb,
                                                 (short)0, acc, false, false);
  }
  return acc;
}

// ---------------- Kernel 2: input projection x_proj = y @ w_in + b_in ----------------
__global__ __launch_bounds__(128)
void k_proj_in(const _Float16* __restrict__ y16, const _Float16* __restrict__ win_t,
               const float* __restrict__ b_in, float* __restrict__ xproj) {
  int lane = threadIdx.x & 31, wv = threadIdx.x >> 5;
  int pix0 = blockIdx.x << 4;
  v8f acc = gemm16x64(y16, win_t, pix0, lane, wv);
  int col = (wv << 4) + (lane & 15);
  float bias = b_in[col];
  #pragma unroll
  for (int i = 0; i < 8; ++i) {
    int pix = pix0 + ((lane & 16) ? (i + 8) : i);
    xproj[(size_t)pix * 64 + col] = acc[i] + bias;
  }
}

// ---------------- Kernel 3: dwconv + LN + GELU + offset/mask proj + softmax ----------------
// One 64-thread block per pixel; thread == channel.
__global__ __launch_bounds__(64)
void k_branch(const float* __restrict__ y32, const float* __restrict__ dw_w,
              const float* __restrict__ dw_b, const float* __restrict__ ln_g,
              const float* __restrict__ ln_b, const float* __restrict__ w_off,
              const float* __restrict__ b_off, const float* __restrict__ w_mask,
              const float* __restrict__ b_mask, float* __restrict__ offs,
              float* __restrict__ maskb) {
  __shared__ float sh[64];
  __shared__ float shm[36];
  int pix = blockIdx.x;
  int n = pix >> 16;
  int hw = pix & 65535;
  int h = hw >> 8, w = hw & 255;
  int c = threadIdx.x;
  float s = dw_b[c];
  #pragma unroll
  for (int kh = 0; kh < 3; ++kh)
    #pragma unroll
    for (int kw = 0; kw < 3; ++kw) {
      int ih = h + kh - 1, iw = w + kw - 1;
      if ((unsigned)ih < OHT && (unsigned)iw < OWD)
        s += y32[(((size_t)n * OHT + ih) * OWD + iw) * 64 + c] * dw_w[(kh * 3 + kw) * 64 + c];
    }
  // LayerNorm over 64 channels (biased variance, eps 1e-5)
  sh[c] = s; __syncthreads();
  for (int off = 32; off > 0; off >>= 1) { if (c < off) sh[c] += sh[c + off]; __syncthreads(); }
  float mean = sh[0] * (1.f / 64.f);
  __syncthreads();
  sh[c] = s * s; __syncthreads();
  for (int off = 32; off > 0; off >>= 1) { if (c < off) sh[c] += sh[c + off]; __syncthreads(); }
  float var = sh[0] * (1.f / 64.f) - mean * mean;
  __syncthreads();
  float v  = (s - mean) * rsqrtf(var + 1e-5f) * ln_g[c] + ln_b[c];
  float xg = 0.5f * v * (1.f + erff(v * 0.70710678118654752f));   // exact GELU
  sh[c] = xg;
  __syncthreads();
  // offset projection: 72 outputs over 64 threads
  float o0 = b_off[c];
  for (int k = 0; k < 64; ++k) o0 += sh[k] * w_off[k * 72 + c];
  offs[(size_t)pix * 72 + c] = o0;
  if (c < 8) {
    int j = c + 64;
    float o1 = b_off[j];
    for (int k = 0; k < 64; ++k) o1 += sh[k] * w_off[k * 72 + j];
    offs[(size_t)pix * 72 + j] = o1;
  }
  // mask projection (36 outputs) + per-group-of-9 softmax
  if (c < 36) {
    float mr = b_mask[c];
    for (int k = 0; k < 64; ++k) mr += sh[k] * w_mask[k * 36 + c];
    shm[c] = mr;
  }
  __syncthreads();
  if (c < 36) {
    int g = c / 9, base = g * 9;
    float mx = shm[base];
    for (int i = 1; i < 9; ++i) mx = fmaxf(mx, shm[base + i]);
    float sum = 0.f;
    for (int i = 0; i < 9; ++i) sum += expf(shm[base + i] - mx);
    maskb[(size_t)pix * 36 + c] = expf(shm[c] - mx) / sum;
  }
}

// ---------------- Kernel 4: deformable bilinear gather (DCNv3 core) ----------------
// One wave per pixel; lane = (group g, channel-pair). acc in f32, store f16.
__global__ __launch_bounds__(256)
void k_dcn(const float* __restrict__ xproj, const float* __restrict__ offs,
           const float* __restrict__ maskb, _Float16* __restrict__ dcn16) {
  int pix  = (blockIdx.x << 3) + (threadIdx.x >> 5);
  int lane = threadIdx.x & 31;
  int g     = lane >> 3;
  int cbase = (g << 4) + ((lane & 7) << 1);
  int n = pix >> 16;
  int hw = pix & 65535;
  int h = hw >> 8, w = hw & 255;
  const float* op = offs  + (size_t)pix * 72 + g * 18;
  const float* mp = maskb + (size_t)pix * 36 + g * 9;
  float acc0 = 0.f, acc1 = 0.f;
  #pragma unroll
  for (int p = 0; p < 9; ++p) {
    int kx = p / 3, ky = p - kx * 3;       // grid: kx outer (x), ky inner (y)
    float ox = op[2 * p], oy = op[2 * p + 1];
    float m  = mp[p];
    // ref math collapses to padded-image coords: ix = w+1+(kx-1)+ox
    float ix = (float)(w + kx) + ox;
    float iy = (float)(h + ky) + oy;
    float x0f = floorf(ix), y0f = floorf(iy);
    float fx = ix - x0f, fy = iy - y0f;
    int x0 = (int)x0f, y0 = (int)y0f;
    float w00 = (1.f - fy) * (1.f - fx) * m;
    float w01 = (1.f - fy) * fx * m;
    float w10 = fy * (1.f - fx) * m;
    float w11 = fy * fx * m;
    #pragma unroll
    for (int cn = 0; cn < 4; ++cn) {
      int yc = y0 + (cn >> 1);
      int xc = x0 + (cn & 1);
      float cw = (cn == 0) ? w00 : (cn == 1) ? w01 : (cn == 2) ? w10 : w11;
      // nonzero only inside the unpadded region (pad ring is zeros)
      if (yc >= 1 && yc <= OHT && xc >= 1 && xc <= OWD) {
        const float* vp = xproj + ((((size_t)n * OHT + (yc - 1)) * OWD) + (xc - 1)) * 64 + cbase;
        acc0 += cw * vp[0];
        acc1 += cw * vp[1];
      }
    }
  }
  dcn16[(size_t)pix * 64 + cbase]     = (_Float16)acc0;
  dcn16[(size_t)pix * 64 + cbase + 1] = (_Float16)acc1;
}

// ---------------- Kernel 5: output projection + NCHW transpose ----------------
__global__ __launch_bounds__(128)
void k_proj_out(const _Float16* __restrict__ dcn16, const _Float16* __restrict__ wout_t,
                const float* __restrict__ b_out, float* __restrict__ out) {
  int lane = threadIdx.x & 31, wv = threadIdx.x >> 5;
  int pix0 = blockIdx.x << 4;
  v8f acc = gemm16x64(dcn16, wout_t, pix0, lane, wv);
  int col = (wv << 4) + (lane & 15);
  float bias = b_out[col];
  #pragma unroll
  for (int i = 0; i < 8; ++i) {
    int pix = pix0 + ((lane & 16) ? (i + 8) : i);
    int nb = pix >> 16;
    int hw = pix & 65535;
    out[((size_t)nb * 64 + col) * 65536 + hw] = acc[i] + bias;   // NCHW
  }
}

extern "C" void kernel_launch(void* const* d_in, const int* in_sizes, int n_in,
                              void* d_out, int out_size, void* d_ws, size_t ws_size,
                              hipStream_t stream) {
  (void)in_sizes; (void)n_in; (void)out_size; (void)ws_size;
  const float* x      = (const float*)d_in[0];
  const float* conv_w = (const float*)d_in[1];
  const float* w_in   = (const float*)d_in[2];
  const float* b_in   = (const float*)d_in[3];
  const float* dw_w   = (const float*)d_in[4];
  const float* dw_b   = (const float*)d_in[5];
  const float* ln_g   = (const float*)d_in[6];
  const float* ln_b   = (const float*)d_in[7];
  const float* w_off  = (const float*)d_in[8];
  const float* b_off  = (const float*)d_in[9];
  const float* w_mask = (const float*)d_in[10];
  const float* b_mask = (const float*)d_in[11];
  const float* w_out  = (const float*)d_in[12];
  const float* b_out  = (const float*)d_in[13];

  char* ws = (char*)d_ws;
  size_t o = 0;
  auto take = [&](size_t bytes) { char* p = ws + o; o += (bytes + 255) & ~(size_t)255; return p; };
  _Float16* cw16   = (_Float16*)take((size_t)64 * KCONV * 2);
  _Float16* win_t  = (_Float16*)take(4096 * 2);
  _Float16* wout_t = (_Float16*)take(4096 * 2);
  float*    y32    = (float*)   take((size_t)NPIX * 64 * 4);
  _Float16* y16    = (_Float16*)take((size_t)NPIX * 64 * 2);
  float*    xproj  = (float*)   take((size_t)NPIX * 64 * 4);
  float*    offsb  = (float*)   take((size_t)NPIX * 72 * 4);
  float*    maskbf = (float*)   take((size_t)NPIX * 36 * 4);
  _Float16* dcn16  = (_Float16*)take((size_t)NPIX * 64 * 2);

  k_prep<<<(64 * KCONV + 8192 + 255) / 256, 256, 0, stream>>>(conv_w, w_in, w_out,
                                                              cw16, win_t, wout_t);
  k_conv<<<8192, 128, 0, stream>>>(x, cw16, y32, y16);
  k_proj_in<<<NPIX / 16, 128, 0, stream>>>(y16, win_t, b_in, xproj);
  k_branch<<<NPIX, 64, 0, stream>>>(y32, dw_w, dw_b, ln_g, ln_b,
                                    w_off, b_off, w_mask, b_mask, offsb, maskbf);
  k_dcn<<<NPIX / 8, 256, 0, stream>>>(xproj, offsb, maskbf, dcn16);
  k_proj_out<<<NPIX / 16, 128, 0, stream>>>(dcn16, wout_t, b_out, (float*)d_out);
}
